// Hawk_78640851189819
// MI455X (gfx1250) — matrix operand order repaired and checked by reading, not compiled, not run on hardware
//
#include <hip/hip_runtime.h>
#include <hip/hip_bf16.h>
#include <math.h>

// Problem constants (match reference)
#define B_  4
#define T_  4096
#define H_  2048
#define R_  2048
#define NB_ 16
#define BW_ 128
#define TW_ 4
#define BT_ (B_*T_)          // 16384
#define R2_ (2*R_)           // 4096

typedef __attribute__((ext_vector_type(16))) __bf16 bf16x16;
typedef __attribute__((ext_vector_type(8)))  float  f32x8;
typedef __attribute__((ext_vector_type(4)))  int    v4i;

union FragU { uint4 u[2]; bf16x16 v; };

// ---------- async global->LDS plumbing (gfx1250) ----------
#if defined(__HIP_DEVICE_COMPILE__) && __has_builtin(__builtin_amdgcn_global_load_async_to_lds_b128)
#define HAVE_ASYNC_LDS 1
#else
#define HAVE_ASYNC_LDS 0
#endif

#if HAVE_ASYNC_LDS
typedef __attribute__((address_space(1))) v4i AS1v4i;
typedef __attribute__((address_space(3))) v4i AS3v4i;
#endif

__device__ __forceinline__ void g2l_b128(const __bf16* g, __bf16* l) {
#if HAVE_ASYNC_LDS
  __builtin_amdgcn_global_load_async_to_lds_b128(
      (AS1v4i*)(g), (AS3v4i*)(l), 0, 0);
#else
  *(uint4*)l = *(const uint4*)g;
#endif
}

__device__ __forceinline__ void wait_async_lds() {
#if HAVE_ASYNC_LDS
#if __has_builtin(__builtin_amdgcn_s_wait_asynccnt)
  __builtin_amdgcn_s_wait_asynccnt(0);
#else
  asm volatile("s_wait_asynccnt 0" ::: "memory");
#endif
#endif
}

__device__ __forceinline__ f32x8 wmma_bf16(const FragU& a, const FragU& b, f32x8 c) {
  return __builtin_amdgcn_wmma_f32_16x16x32_bf16(
      /*neg_a=*/false, a.v, /*neg_b=*/false, b.v,
      /*c_mod=*/(short)0, c, /*reuse_a=*/false, /*reuse_b=*/false);
}

// ---------------- prep kernels ----------------

__global__ void k_cvt_bf16(const float* __restrict__ s, __bf16* __restrict__ d, size_t n) {
  size_t i = (size_t)blockIdx.x * blockDim.x + threadIdx.x;
  if (i < n) d[i] = (__bf16)s[i];
}

// wi/wa: [NB][d][e] (contract over d) -> wT: [NB][e][d] so K is contiguous
__global__ void k_transpose_wb(const float* __restrict__ w, __bf16* __restrict__ wT) {
  size_t i = (size_t)blockIdx.x * blockDim.x + threadIdx.x;
  if (i >= (size_t)NB_ * BW_ * BW_) return;
  int nb  = (int)(i / (BW_ * BW_));
  int rem = (int)(i % (BW_ * BW_));
  int e = rem / BW_;
  int d = rem % BW_;
  wT[i] = (__bf16)w[(size_t)nb * BW_ * BW_ + (size_t)d * BW_ + e];
}

// ---------------- block-tiled WMMA GEMM ----------------
// C[m,n] = sum_k A[m,k] * W[n,k]   (A,W bf16 row-major, K contiguous)
// Block: 256 threads = 8 wave32; 128x128 C tile per block.
// Wave (wm = wave>>2, wn = wave&3) computes a 64x32 patch = 4x2 WMMA tiles.
// K-slices (128x32 bf16) of A and W staged in LDS, double buffered,
// loaded with async global->LDS; rows padded to 40 elems (20 dwords,
// conflict-free for the b128 fragment reads, 16B aligned).
// Fragment layouts used here (wave32, contiguous k order for A/B):
//   A 16x32: lane -> row = lane&15; k = 16*(lane>>4)+[0,16)
//   B 32x16: lane -> col = lane&15; k = 16*(lane>>4)+[0,16)
//   C 16x16: VGPR v at (m = 8*(lane>>4)+v, n = lane&15)
#define KS 32
#define LDSW 40   // padded LDS row stride (elements)

__global__ void __launch_bounds__(256)
k_gemm_bf16(const __bf16* __restrict__ A, const __bf16* __restrict__ W,
            void* __restrict__ Cv,
            int M, int N, int K,
            long lda, long ldw, long ldc,
            long sA, long sW, long sC, int outBf16)
{
  __shared__ __align__(16) __bf16 Abuf[2][128 * LDSW];
  __shared__ __align__(16) __bf16 Bbuf[2][128 * LDSW];

  const int tid  = threadIdx.x;
  const int lane = tid & 31;
  const int wave = tid >> 5;
  const int wm   = wave >> 2;      // 0..1 : 64-row band
  const int wn   = wave & 3;       // 0..3 : 32-col band
  const int row  = lane & 15;
  const int hi   = lane >> 4;

  const long zb    = blockIdx.z;
  const int  mBase = blockIdx.y * 128;
  const int  nBase = blockIdx.x * 128;
  const __bf16* Ab = A + zb * sA + (long)mBase * lda;
  const __bf16* Wb = W + zb * sW + (long)nBase * ldw;

  f32x8 acc[4][2] = {};

  // --- stage K-slice 0 into buffer 0 ---
  #pragma unroll
  for (int c = 0; c < 2; ++c) {
    int q  = tid + c * 256;        // 512 x 16B chunks per matrix slice
    int r  = q >> 2;               // tile row 0..127
    int ko = (q & 3) * 8;          // k offset 0/8/16/24
    g2l_b128(Ab + (long)r * lda + ko, &Abuf[0][r * LDSW + ko]);
    g2l_b128(Wb + (long)r * ldw + ko, &Bbuf[0][r * LDSW + ko]);
  }
  wait_async_lds();
  __syncthreads();

  const int nk = K / KS;
  for (int s = 0; s < nk; ++s) {
    const int buf = s & 1;
    if (s + 1 < nk) {
      const int kk = (s + 1) * KS;
      #pragma unroll
      for (int c = 0; c < 2; ++c) {
        int q  = tid + c * 256;
        int r  = q >> 2;
        int ko = (q & 3) * 8;
        g2l_b128(Ab + (long)r * lda + kk + ko, &Abuf[buf ^ 1][r * LDSW + ko]);
        g2l_b128(Wb + (long)r * ldw + kk + ko, &Bbuf[buf ^ 1][r * LDSW + ko]);
      }
    }

    // B fragments (shared across the 4 M-subtiles)
    FragU fb0, fb1;
    {
      const __bf16* bp0 = &Bbuf[buf][(wn * 32 +  0 + row) * LDSW + hi * 16];
      const __bf16* bp1 = &Bbuf[buf][(wn * 32 + 16 + row) * LDSW + hi * 16];
      fb0.u[0] = *(const uint4*)(bp0);
      fb0.u[1] = *(const uint4*)(bp0 + 8);
      fb1.u[0] = *(const uint4*)(bp1);
      fb1.u[1] = *(const uint4*)(bp1 + 8);
    }
    #pragma unroll
    for (int mi = 0; mi < 4; ++mi) {
      const __bf16* ap = &Abuf[buf][(wm * 64 + mi * 16 + row) * LDSW + hi * 16];
      FragU fa;
      fa.u[0] = *(const uint4*)(ap);
      fa.u[1] = *(const uint4*)(ap + 8);
      acc[mi][0] = wmma_bf16(fa, fb0, acc[mi][0]);
      acc[mi][1] = wmma_bf16(fa, fb1, acc[mi][1]);
    }

    wait_async_lds();   // next buffer's async stores complete
    __syncthreads();    // visible to all waves
  }

  // --- store C ---
  const int col0 = nBase + wn * 32 + row;
  #pragma unroll
  for (int mi = 0; mi < 4; ++mi) {
    const int m0 = mBase + wm * 64 + mi * 16 + hi * 8;
    #pragma unroll
    for (int ni = 0; ni < 2; ++ni) {
      const int n = col0 + ni * 16;
      if (outBf16) {
        __bf16* C = (__bf16*)Cv;
        #pragma unroll
        for (int v = 0; v < 8; ++v)
          C[zb * sC + (long)(m0 + v) * ldc + n] = (__bf16)acc[mi][ni][v];
      } else {
        float* C = (float*)Cv;
        #pragma unroll
        for (int v = 0; v < 8; ++v)
          C[zb * sC + (long)(m0 + v) * ldc + n] = acc[mi][ni][v];
      }
    }
  }
}

// ---------------- causal depthwise conv (TW=4) ----------------
__global__ void k_conv(const __bf16* __restrict__ xy, const float* __restrict__ cw,
                       const float* __restrict__ cb, __bf16* __restrict__ xc)
{
  size_t i = (size_t)blockIdx.x * blockDim.x + threadIdx.x;
  if (i >= (size_t)BT_ * R_) return;
  int    r  = (int)(i % R_);
  size_t bt = i / R_;
  int    t  = (int)(bt % T_);
  size_t b0 = bt - (size_t)t;              // b*T
  float acc = cb[r];
  #pragma unroll
  for (int k = 0; k < TW_; ++k) {
    int tt = t - (TW_ - 1) + k;
    if (tt >= 0)
      acc += cw[k * R_ + r] * (float)xy[(b0 + (size_t)tt) * R2_ + r];
  }
  xc[i] = (__bf16)acc;
}

// ---------------- RG-LRU gate math ----------------
__global__ void k_gates(const __bf16* __restrict__ xc, const float* __restrict__ bi,
                        const float* __restrict__ ba, const float* __restrict__ ap,
                        float* __restrict__ g1, float* __restrict__ g2)
{
  size_t i = (size_t)blockIdx.x * blockDim.x + threadIdx.x;
  if (i >= (size_t)BT_ * R_) return;
  int r = (int)(i % R_);
  float gx = 1.f / (1.f + __expf(-(g1[i] + bi[r])));
  float ga = 1.f / (1.f + __expf(-(g2[i] + ba[r])));
  float sp = log1pf(__expf(ap[r]));                    // softplus(a_param)
  float log_a = -8.f * ga * sp;
  float a = __expf(log_a);
  float mult = sqrtf(fmaxf(1.f - __expf(2.f * log_a), 0.f));  // sqrt_bound fwd
  float nx = (float)xc[i] * gx * mult;
  g2[i] = a;
  g1[i] = nx;
}

// ---------------- linear recurrence scan ----------------
__global__ void k_scan(const float* __restrict__ a, float* __restrict__ h)
{
  int c = blockIdx.x * blockDim.x + threadIdx.x;
  if (c >= B_ * R_) return;
  int b = c / R_;
  int r = c % R_;
  size_t base = (size_t)b * T_ * R_ + (size_t)r;
  float hp = 0.f;
  for (int t = 0; t < T_; ++t) {
    size_t i = base + (size_t)t * R_;
    hp = a[i] * hp + h[i];
    h[i] = hp;
  }
}

// ---------------- z = gelu(y) * h (exact gelu) ----------------
__global__ void k_zmul(const __bf16* __restrict__ xy, const float* __restrict__ h,
                       __bf16* __restrict__ z)
{
  size_t i = (size_t)blockIdx.x * blockDim.x + threadIdx.x;
  if (i >= (size_t)BT_ * R_) return;
  int    r  = (int)(i % R_);
  size_t bt = i / R_;
  float y = (float)xy[bt * R2_ + R_ + r];
  float g = 0.5f * y * (1.f + erff(y * 0.70710678118654752f));
  z[i] = (__bf16)(g * h[i]);
}

// ---------------- host launch ----------------
extern "C" void kernel_launch(void* const* d_in, const int* in_sizes, int n_in,
                              void* d_out, int out_size, void* d_ws, size_t ws_size,
                              hipStream_t stream) {
  (void)in_sizes; (void)n_in; (void)out_size; (void)ws_size;
  const float* x    = (const float*)d_in[0];   // [B,T,H]
  const float* Wxy  = (const float*)d_in[1];   // [2R,H]
  const float* wi   = (const float*)d_in[2];   // [NB,BW,BW]
  const float* bi   = (const float*)d_in[3];   // [NB,BW] (flat R)
  const float* wa   = (const float*)d_in[4];
  const float* ba   = (const float*)d_in[5];
  const float* ap   = (const float*)d_in[6];   // [R]
  const float* cw   = (const float*)d_in[7];   // [TW,R]
  const float* cb   = (const float*)d_in[8];   // [R]
  const float* Wout = (const float*)d_in[9];   // [H,R]
  float* out = (float*)d_out;                  // [B,T,H] f32

  char* ws = (char*)d_ws;
  size_t off = 0;
  auto alloc = [&](size_t bytes) -> void* {
    off = (off + 255) & ~(size_t)255;
    void* p = ws + off;
    off += bytes;
    return p;
  };

  __bf16* xb  = (__bf16*)alloc((size_t)BT_ * H_ * 2);        // x bf16
  __bf16* w1  = (__bf16*)alloc((size_t)R2_ * H_ * 2);        // W_xy bf16
  __bf16* wiT = (__bf16*)alloc((size_t)NB_ * BW_ * BW_ * 2); // wi^T bf16
  __bf16* waT = (__bf16*)alloc((size_t)NB_ * BW_ * BW_ * 2); // wa^T bf16
  __bf16* wo  = (__bf16*)alloc((size_t)H_ * R_ * 2);         // W_out bf16
  __bf16* xyb = (__bf16*)alloc((size_t)BT_ * R2_ * 2);       // xy = x@W_xy^T
  __bf16* xc  = (__bf16*)alloc((size_t)BT_ * R_ * 2);        // conv out; later z
  float*  g1  = (float*)alloc((size_t)BT_ * R_ * 4);         // gxl -> nx -> h
  float*  g2  = (float*)alloc((size_t)BT_ * R_ * 4);         // gal -> a

  const int TPB = 256;
  auto grid1 = [&](size_t n) { return dim3((unsigned)((n + TPB - 1) / TPB)); };

  // prep: conversions + block-weight transpose
  k_cvt_bf16<<<grid1((size_t)BT_ * H_), TPB, 0, stream>>>(x, xb, (size_t)BT_ * H_);
  k_cvt_bf16<<<grid1((size_t)R2_ * H_), TPB, 0, stream>>>(Wxy, w1, (size_t)R2_ * H_);
  k_transpose_wb<<<grid1((size_t)NB_ * BW_ * BW_), TPB, 0, stream>>>(wi, wiT);
  k_transpose_wb<<<grid1((size_t)NB_ * BW_ * BW_), TPB, 0, stream>>>(wa, waT);
  k_cvt_bf16<<<grid1((size_t)H_ * R_), TPB, 0, stream>>>(Wout, wo, (size_t)H_ * R_);

  // GEMM1: xy[BT,2R] = x[BT,H] @ W_xy[2R,H]^T  (bf16 out)
  k_gemm_bf16<<<dim3(R2_ / 128, BT_ / 128, 1), 256, 0, stream>>>(
      xb, w1, xyb, BT_, R2_, H_, H_, H_, R2_, 0, 0, 0, /*outBf16=*/1);

  // causal conv on xr half
  k_conv<<<grid1((size_t)BT_ * R_), TPB, 0, stream>>>(xyb, cw, cb, xc);

  // batched block-diag gate GEMMs (grid.z = NB blocks of 128x128)
  k_gemm_bf16<<<dim3(BW_ / 128, BT_ / 128, NB_), 256, 0, stream>>>(
      xc, wiT, g1, BT_, BW_, BW_, R_, BW_, R_,
      /*sA=*/BW_, /*sW=*/(long)BW_ * BW_, /*sC=*/BW_, /*outBf16=*/0);
  k_gemm_bf16<<<dim3(BW_ / 128, BT_ / 128, NB_), 256, 0, stream>>>(
      xc, waT, g2, BT_, BW_, BW_, R_, BW_, R_,
      /*sA=*/BW_, /*sW=*/(long)BW_ * BW_, /*sC=*/BW_, /*outBf16=*/0);

  // gate math: g1 <- nx, g2 <- a
  k_gates<<<grid1((size_t)BT_ * R_), TPB, 0, stream>>>(xc, bi, ba, ap, g1, g2);

  // recurrence: h in place of nx (g1)
  k_scan<<<dim3((B_ * R_) / TPB), TPB, 0, stream>>>(g2, g1);

  // z = gelu(y) * h  (bf16, reuse xc)
  k_zmul<<<grid1((size_t)BT_ * R_), TPB, 0, stream>>>(xyb, g1, xc);

  // GEMM3: out[BT,H] = z[BT,R] @ W_out[H,R]^T  (f32 out)
  k_gemm_bf16<<<dim3(H_ / 128, BT_ / 128, 1), 256, 0, stream>>>(
      xc, wo, out, BT_, H_, R_, R_, R_, H_, 0, 0, 0, /*outBf16=*/0);
}
